// SC2HybridBridge_57045755625698
// MI455X (gfx1250) — compile-verified, matrix operand order checked by reading, not run on hardware
//
#include <hip/hip_runtime.h>
#include <stdint.h>

#define L_SEQ  2000
#define BSZ_   8
#define D_MOD  128
#define DIN_   256
#define NST    64
#define KCONV  4
#define NHEADS 4
#define BL     (BSZ_ * L_SEQ)   // 16000 rows

typedef __attribute__((ext_vector_type(16))) __bf16 v16bf;
typedef __attribute__((ext_vector_type(8)))  __bf16 v8bf;
typedef __attribute__((ext_vector_type(8)))  float  v8f;

#if defined(__has_builtin)
#if __has_builtin(__builtin_amdgcn_global_load_async_to_lds_b128) && \
    __has_builtin(__builtin_amdgcn_s_wait_asynccnt)
#define ASYNC_LDS 1
#endif
#endif

#ifdef ASYNC_LDS
typedef int v4i __attribute__((vector_size(16)));
typedef __attribute__((address_space(1))) v4i gv4i;   // global int4
typedef __attribute__((address_space(3))) v4i lv4i;   // LDS int4
#endif

// ---------------------------------------------------------------------------
// Embedding: h[b,l,d] = x[b,l]*Win_b[d] + bin_b[d] + gene_emb[l,d] + mod_emb[d]
// ---------------------------------------------------------------------------
__global__ void k_embed(const float* __restrict__ x, const float* __restrict__ wb,
                        const float* __restrict__ bb, const float* __restrict__ ge,
                        const float* __restrict__ me, float* __restrict__ h)
{
    int idx = blockIdx.x * blockDim.x + threadIdx.x;
    if (idx >= BL * D_MOD) return;
    int row = idx >> 7;         // token row (b*L + l)
    int d   = idx & 127;
    int l   = row % L_SEQ;
    h[idx] = x[row] * wb[d] + bb[d] + ge[l * D_MOD + d] + me[d];
}

// ---------------------------------------------------------------------------
// LayerNorm over D=128, one wave per row, bf16 output (feeds WMMA GEMMs)
// ---------------------------------------------------------------------------
__global__ __launch_bounds__(256) void k_ln_bf16(const float* __restrict__ xin,
        const float* __restrict__ g, const float* __restrict__ bta,
        __bf16* __restrict__ out, int rows)
{
    int wid  = (blockIdx.x * blockDim.x + threadIdx.x) >> 5;
    int lane = threadIdx.x & 31;
    if (wid >= rows) return;
    const float* xr = xin + (size_t)wid * D_MOD;
    float v[4]; float s = 0.f;
    #pragma unroll
    for (int i = 0; i < 4; i++) { v[i] = xr[lane * 4 + i]; s += v[i]; }
    #pragma unroll
    for (int off = 16; off > 0; off >>= 1) s += __shfl_xor(s, off, 32);
    float mu = s * (1.0f / D_MOD);
    float var = 0.f;
    #pragma unroll
    for (int i = 0; i < 4; i++) { float dd = v[i] - mu; var += dd * dd; }
    #pragma unroll
    for (int off = 16; off > 0; off >>= 1) var += __shfl_xor(var, off, 32);
    float inv = rsqrtf(var * (1.0f / D_MOD) + 1e-5f);
    __bf16* orow = out + (size_t)wid * D_MOD;
    #pragma unroll
    for (int i = 0; i < 4; i++) {
        int d = lane * 4 + i;
        orow[d] = (__bf16)((v[i] - mu) * inv * g[d] + bta[d]);
    }
}

// ---------------------------------------------------------------------------
// Weight convert f32 [N][K] -> bf16 [Npad][K], zero padding rows >= N
// ---------------------------------------------------------------------------
__global__ void k_cvtw(const float* __restrict__ W, __bf16* __restrict__ out,
                       int N, int Npad, int K)
{
    int idx = blockIdx.x * blockDim.x + threadIdx.x;
    if (idx >= Npad * K) return;
    int n = idx / K;
    out[idx] = (n < N) ? (__bf16)W[idx] : (__bf16)0.0f;
}

// Generic f32 -> bf16 convert
__global__ void k_cvt(const float* __restrict__ src, __bf16* __restrict__ dst, int n)
{
    int idx = blockIdx.x * blockDim.x + threadIdx.x;
    if (idx < n) dst[idx] = (__bf16)src[idx];
}

// ---------------------------------------------------------------------------
// bf16 WMMA GEMM: C[M][ldc] = A[M][K] * B[N][K]^T (+bias)(+C).
// Block = 8 waves = 128x64 macro-tile. The 64xK bf16 B-panel is staged into
// LDS once per block (async global->LDS when available) and shared by all
// 8 waves; each wave then runs 16x64 WMMA tiles out of LDS fragments.
// ---------------------------------------------------------------------------
__global__ __launch_bounds__(256) void k_gemm(const __bf16* __restrict__ A,
        const __bf16* __restrict__ B, float* __restrict__ C,
        const float* __restrict__ bias, int M, int N, int K, int ldc, int accum)
{
    __shared__ __bf16 Bs[64 * 264];          // rows padded to K+8 (K<=256)
    const int w    = threadIdx.x >> 5;
    const int lane = threadIdx.x & 31;
    const int m0 = (blockIdx.x * 8 + w) * 16;
    const int n0 = blockIdx.y * 64;
    const int ldb = K + 8;

    // ---- cooperative B panel staging: 64 rows x K bf16 ----
    {
        const int chunksPerRow = K >> 3;                // 16B chunks per row
        const int total = 64 * chunksPerRow;
        for (int e = threadIdx.x; e < total; e += 256) {
            int row = e / chunksPerRow;
            int c8  = e - row * chunksPerRow;
            const __bf16* src = B + (size_t)(n0 + row) * K + c8 * 8;
            __bf16*       dst = Bs + row * ldb + c8 * 8;
#ifdef ASYNC_LDS
            __builtin_amdgcn_global_load_async_to_lds_b128(
                (gv4i*)(uintptr_t)src, (lv4i*)(uintptr_t)dst, 0, 0);
#else
            *(v8bf*)dst = *(const v8bf*)src;
#endif
        }
#ifdef ASYNC_LDS
        __builtin_amdgcn_s_wait_asynccnt(0);
#endif
    }
    __syncthreads();

    const bool active = (m0 < M);
    const int r    = lane & 15;
    const int hlf  = lane >> 4;
    v8f acc[4] = { {0.f}, {0.f}, {0.f}, {0.f} };
    if (active) {
        const __bf16* arow = A + (size_t)(m0 + r) * K + hlf * 8;
        for (int kk = 0; kk < K; kk += 32) {
            v16bf a;
            {
                v8bf lo = *(const v8bf*)(arow + kk);
                v8bf hi = *(const v8bf*)(arow + kk + 16);
                #pragma unroll
                for (int i = 0; i < 8; i++) { a[i] = lo[i]; a[i + 8] = hi[i]; }
            }
            if (kk + 32 < K) __builtin_prefetch(arow + kk + 32, 0, 3);
            #pragma unroll
            for (int j = 0; j < 4; j++) {
                const __bf16* brow = Bs + (j * 16 + r) * ldb + hlf * 8 + kk;
                v8bf lo = *(const v8bf*)brow;
                v8bf hi = *(const v8bf*)(brow + 16);
                v16bf bb;
                #pragma unroll
                for (int i = 0; i < 8; i++) { bb[i] = lo[i]; bb[i + 8] = hi[i]; }
                acc[j] = __builtin_amdgcn_wmma_f32_16x16x32_bf16(
                             false, a, false, bb, (short)0, acc[j], false, false);
            }
        }
        #pragma unroll
        for (int j = 0; j < 4; j++) {
            const int col = n0 + j * 16 + r;
            const float bv = bias ? bias[col] : 0.0f;
            #pragma unroll
            for (int i = 0; i < 8; i++) {
                const int row = m0 + hlf * 8 + i;
                const size_t idx = (size_t)row * ldc + col;
                float v = acc[j][i] + bv;
                if (accum) v += C[idx];
                C[idx] = v;
            }
        }
    }
}

// ---------------------------------------------------------------------------
// Depthwise causal conv (K=4) + bias + SiLU over xz[:, :256]; f32 + bf16 out
// ---------------------------------------------------------------------------
__global__ void k_conv(const float* __restrict__ xz, const float* __restrict__ cw,
                       const float* __restrict__ cb, float* __restrict__ xc,
                       __bf16* __restrict__ xcbf)
{
    int idx = blockIdx.x * blockDim.x + threadIdx.x;
    if (idx >= BL * DIN_) return;
    int row = idx >> 8;
    int d   = idx & 255;
    int b = row / L_SEQ, l = row % L_SEQ;
    float acc = cb[d];
    #pragma unroll
    for (int j = 0; j < KCONV; j++) {
        int lj = l - (KCONV - 1) + j;
        if (lj >= 0)
            acc += cw[d * KCONV + j] * xz[((size_t)(b * L_SEQ + lj)) * 512 + d];
    }
    float r = acc / (1.f + __expf(-acc));   // SiLU
    xc[idx]   = r;
    xcbf[idx] = (__bf16)r;
}

// ---------------------------------------------------------------------------
// dt = softplus(dtr @ Wdt^T + bdt); dtr = xdb[:, :8] (K=8, VALU)
// ---------------------------------------------------------------------------
__global__ void k_dt(const float* __restrict__ xdb, const float* __restrict__ Wdt,
                     const float* __restrict__ bdt, float* __restrict__ out)
{
    int idx = blockIdx.x * blockDim.x + threadIdx.x;
    if (idx >= BL * DIN_) return;
    int row = idx >> 8;
    int d   = idx & 255;
    float acc = bdt[d];
    #pragma unroll
    for (int r = 0; r < 8; r++)
        acc += xdb[(size_t)row * 192 + r] * Wdt[d * 8 + r];
    out[idx] = (acc > 20.f) ? acc : log1pf(__expf(acc));
}

// ---------------------------------------------------------------------------
// Selective scan: one wave per (b,d), 2 states per lane, shfl reduction for y
// ---------------------------------------------------------------------------
__global__ __launch_bounds__(256) void k_scan(const float* __restrict__ dt,
        const float* __restrict__ xc, const float* __restrict__ xdb,
        const float* __restrict__ A_log, float* __restrict__ ys)
{
    int wg   = blockIdx.x * 8 + (threadIdx.x >> 5);
    int lane = threadIdx.x & 31;
    int b = wg >> 8;
    int d = wg & 255;
    int n1 = lane, n2 = lane + 32;
    float A1 = -__expf(A_log[d * NST + n1]);
    float A2 = -__expf(A_log[d * NST + n2]);
    float h1 = 0.f, h2 = 0.f;
    size_t rbase = (size_t)b * L_SEQ;
    for (int l = 0; l < L_SEQ; l++) {
        size_t row = rbase + l;
        float dtv = dt[row * DIN_ + d];
        float xv  = xc[row * DIN_ + d];
        const float* br = xdb + row * 192 + 8;    // B: cols 8..71
        const float* cr = xdb + row * 192 + 72;   // C: cols 72..135
        float u = dtv * xv;
        h1 = h1 * __expf(dtv * A1) + u * br[n1];
        h2 = h2 * __expf(dtv * A2) + u * br[n2];
        float y = h1 * cr[n1] + h2 * cr[n2];
        #pragma unroll
        for (int off = 16; off > 0; off >>= 1) y += __shfl_xor(y, off, 32);
        if (lane == 0) ys[row * DIN_ + d] = y;
    }
}

// ---------------------------------------------------------------------------
// Gate: out = (ys + xc*Dp) * SiLU(z),  z = xz[:, 256+d]; bf16 out for Wout GEMM
// ---------------------------------------------------------------------------
__global__ void k_gate(const float* __restrict__ ys, const float* __restrict__ xc,
                       const float* __restrict__ Dp, const float* __restrict__ xz,
                       __bf16* __restrict__ out)
{
    int idx = blockIdx.x * blockDim.x + threadIdx.x;
    if (idx >= BL * DIN_) return;
    int row = idx >> 8;
    int d   = idx & 255;
    float y = ys[idx] + xc[idx] * Dp[d];
    float z = xz[(size_t)row * 512 + 256 + d];
    out[idx] = (__bf16)(y * (z / (1.f + __expf(-z))));
}

// ---------------------------------------------------------------------------
// Flash-style attention, all-WMMA: per (b,head,16-query tile), one wave.
// For each 32-key chunk:
//   S^T = K_chunk x Q^T        (2 WMMAs; C-layout rows = key -> LDS [key][q])
//   online softmax on LDS      (lanes 0..15 own one query row each)
//   P read back from LDS in A-operand layout (row=query, K=key) -- free
//   transpose through LDS; O += P x V (2 WMMAs, hd split 16+16, V gathered
//   K-major). Tail keys (2000 % 32 = 16) masked by zeroing P.
// ---------------------------------------------------------------------------
__global__ __launch_bounds__(32) void k_attn(const float* __restrict__ qkv,
                                             float* __restrict__ o)
{
    const int QT = L_SEQ / 16;                // 125 query tiles
    int blk = blockIdx.x;
    int qt  = blk % QT;
    int tmp = blk / QT;
    int hh  = tmp % NHEADS;
    int b   = tmp / NHEADS;
    int lane = threadIdx.x;
    int r    = lane & 15;
    int hlf  = lane >> 4;
    int kb   = hlf * 8;
    int q0   = qt * 16;
    const float* base = qkv + (size_t)b * L_SEQ * 384;

    // Q as B-operand for S^T = K x Q^T (N = query, K = d), loaded once
    v16bf bq;
    {
        const float* qrow = base + (size_t)(q0 + r) * 384 + hh * 32;
        #pragma unroll
        for (int i = 0; i < 8; i++) { bq[i]     = (__bf16)qrow[kb + i];
                                      bq[i + 8] = (__bf16)qrow[kb + 16 + i]; }
    }

    __shared__ float S[32][17];               // [key][query]
    __shared__ float F[16];
    __shared__ float Lsum[16];
    float m_r = -3.0e38f, l_r = 0.f;          // lanes 0..15 own query = lane
    v8f o1 = {0.f}, o2 = {0.f};               // O cols 0..15 / 16..31 of head

    const int NCH = (L_SEQ + 31) / 32;        // 63 chunks
    for (int c = 0; c < NCH; c++) {
        int k0 = c * 32;
        // ---- S^T chunk: two WMMAs, A = K rows (clamped for safe tail loads)
        #pragma unroll
        for (int t2 = 0; t2 < 2; t2++) {
            int kr  = k0 + t2 * 16 + r;
            int krc = (kr < L_SEQ) ? kr : (L_SEQ - 1);
            const float* krow = base + (size_t)krc * 384 + 128 + hh * 32;
            v16bf ak;
            #pragma unroll
            for (int i = 0; i < 8; i++) { ak[i]     = (__bf16)krow[kb + i];
                                          ak[i + 8] = (__bf16)krow[kb + 16 + i]; }
            v8f cs = {0.f};
            cs = __builtin_amdgcn_wmma_f32_16x16x32_bf16(
                     false, ak, false, bq, (short)0, cs, false, false);
            #pragma unroll
            for (int i = 0; i < 8; i++)
                S[t2 * 16 + i + hlf * 8][r] = cs[i] * 0.17677669529663687f;
        }
        __syncthreads();
        // ---- online softmax (row = query owned by lanes 0..15)
        int nvalid = (L_SEQ - k0 < 32) ? (L_SEQ - k0) : 32;
        if (lane < 16) {
            float mt = -3.0e38f;
            for (int j = 0; j < nvalid; j++) mt = fmaxf(mt, S[j][lane]);
            float nm = fmaxf(m_r, mt);
            float f  = __expf(m_r - nm);
            float s  = 0.f;
            for (int j = 0; j < 32; j++) {
                float p = (j < nvalid) ? __expf(S[j][lane] - nm) : 0.f;
                S[j][lane] = p;
                s += p;
            }
            l_r = l_r * f + s;
            m_r = nm;
            F[lane] = f;
        }
        __syncthreads();
        // ---- P as A-operand straight out of LDS (row=query, K=key)
        v16bf ap;
        #pragma unroll
        for (int i = 0; i < 8; i++) { ap[i]     = (__bf16)S[kb + i][r];
                                      ap[i + 8] = (__bf16)S[kb + 16 + i][r]; }
        // rescale O by per-query factor
        #pragma unroll
        for (int i = 0; i < 8; i++) { float f = F[i + hlf * 8]; o1[i] *= f; o2[i] *= f; }
        // ---- O += P x V (two hd halves); V gathered K-major (B operand)
        #pragma unroll
        for (int t2 = 0; t2 < 2; t2++) {
            v16bf bv;
            #pragma unroll
            for (int i = 0; i < 8; i++) {
                int kr   = k0 + kb + i;       int krc  = (kr  < L_SEQ) ? kr  : (L_SEQ - 1);
                int kr2  = k0 + kb + 16 + i;  int krc2 = (kr2 < L_SEQ) ? kr2 : (L_SEQ - 1);
                bv[i]     = (__bf16)base[(size_t)krc  * 384 + 256 + hh * 32 + t2 * 16 + r];
                bv[i + 8] = (__bf16)base[(size_t)krc2 * 384 + 256 + hh * 32 + t2 * 16 + r];
            }
            if (t2 == 0)
                o1 = __builtin_amdgcn_wmma_f32_16x16x32_bf16(
                         false, ap, false, bv, (short)0, o1, false, false);
            else
                o2 = __builtin_amdgcn_wmma_f32_16x16x32_bf16(
                         false, ap, false, bv, (short)0, o2, false, false);
        }
        __syncthreads();
    }
    if (lane < 16) Lsum[lane] = l_r;
    __syncthreads();
    #pragma unroll
    for (int i = 0; i < 8; i++) {
        int q = q0 + i + hlf * 8;
        float inv = 1.0f / Lsum[i + hlf * 8];
        float* orow = o + ((size_t)(b * L_SEQ + q)) * 128 + hh * 32;
        orow[r]      = o1[i] * inv;
        orow[16 + r] = o2[i] * inv;
    }
}

// ---------------------------------------------------------------------------
// Final LN + head: out[b,l] = LN(h) . Wh + bh ; one wave per row
// ---------------------------------------------------------------------------
__global__ __launch_bounds__(256) void k_final(const float* __restrict__ hin,
        const float* __restrict__ g, const float* __restrict__ bta,
        const float* __restrict__ Wh, const float* __restrict__ bh,
        float* __restrict__ out, int rows)
{
    int wid  = (blockIdx.x * blockDim.x + threadIdx.x) >> 5;
    int lane = threadIdx.x & 31;
    if (wid >= rows) return;
    const float* xr = hin + (size_t)wid * D_MOD;
    float v[4]; float s = 0.f;
    #pragma unroll
    for (int i = 0; i < 4; i++) { v[i] = xr[lane * 4 + i]; s += v[i]; }
    #pragma unroll
    for (int off = 16; off > 0; off >>= 1) s += __shfl_xor(s, off, 32);
    float mu = s * (1.0f / D_MOD);
    float var = 0.f;
    #pragma unroll
    for (int i = 0; i < 4; i++) { float dd = v[i] - mu; var += dd * dd; }
    #pragma unroll
    for (int off = 16; off > 0; off >>= 1) var += __shfl_xor(var, off, 32);
    float inv = rsqrtf(var * (1.0f / D_MOD) + 1e-5f);
    float dot = 0.f;
    #pragma unroll
    for (int i = 0; i < 4; i++) {
        int d = lane * 4 + i;
        dot += ((v[i] - mu) * inv * g[d] + bta[d]) * Wh[d];
    }
    #pragma unroll
    for (int off = 16; off > 0; off >>= 1) dot += __shfl_xor(dot, off, 32);
    if (lane == 0) out[wid] = dot + bh[0];
}

// ---------------------------------------------------------------------------
// Host launch
// ---------------------------------------------------------------------------
extern "C" void kernel_launch(void* const* d_in, const int* in_sizes, int n_in,
                              void* d_out, int out_size, void* d_ws, size_t ws_size,
                              hipStream_t stream)
{
    (void)in_sizes; (void)n_in; (void)out_size; (void)ws_size;
    // Input order (recursive dict insertion order of setup_inputs):
    // 0:x 1:Win_b 2:bin_b 3:gene_emb 4:mod_emb
    // 5+i*11: mamba[i] = {ln_g, ln_b, Win, conv_w, conv_b, Wx, Wdt, bdt, A_log, Dp, Wout}
    // 71+j*6: attn[j]  = {ln_g, ln_b, Wqkv, bqkv, Wo, bo}
    // 83:fn_g 84:fn_b 85:Wh 86:bh
    const float* x     = (const float*)d_in[0];

    float*  h    = (float*)d_ws;                                   // BL*128 f32
    __bf16* wbuf = (__bf16*)((char*)d_ws + (size_t)BL * 128 * 4);  // 65536 bf16
    __bf16* xbf  = (__bf16*)((char*)wbuf + (size_t)65536 * 2);     // BL*256 bf16
    float*  t0   = (float*)((char*)xbf + (size_t)BL * 256 * 2);    // BL*512  (xz / qkv)
    float*  t1   = t0 + (size_t)BL * 512;                          // BL*256  (xc)
    float*  t2   = t1 + (size_t)BL * 256;                          // BL*192  (xdb padded)
    float*  t3   = t2 + (size_t)BL * 192;                          // BL*256  (dt)
    float*  t4   = t3 + (size_t)BL * 256;                          // BL*256  (ys / o)

    k_embed<<<(BL * 128 + 255) / 256, 256, 0, stream>>>(
        x, (const float*)d_in[1], (const float*)d_in[2],
        (const float*)d_in[3], (const float*)d_in[4], h);

    int ai = 0;
    for (int i = 0; i < 6; i++) {
        const int mb = 5 + i * 11;
        k_ln_bf16<<<2000, 256, 0, stream>>>(h, (const float*)d_in[mb + 0],
                                            (const float*)d_in[mb + 1], xbf, BL);
        k_cvtw<<<(512 * 128 + 255) / 256, 256, 0, stream>>>(
            (const float*)d_in[mb + 2], wbuf, 512, 512, 128);
        k_gemm<<<dim3(125, 8), 256, 0, stream>>>(xbf, wbuf, t0, nullptr,
                                                 BL, 512, 128, 512, 0);
        k_conv<<<(BL * 256 + 255) / 256, 256, 0, stream>>>(
            t0, (const float*)d_in[mb + 3], (const float*)d_in[mb + 4], t1, xbf);
        k_cvtw<<<(192 * 256 + 255) / 256, 256, 0, stream>>>(
            (const float*)d_in[mb + 5], wbuf, 136, 192, 256);
        k_gemm<<<dim3(125, 3), 256, 0, stream>>>(xbf, wbuf, t2, nullptr,
                                                 BL, 192, 256, 192, 0);
        k_dt<<<(BL * 256 + 255) / 256, 256, 0, stream>>>(
            t2, (const float*)d_in[mb + 6], (const float*)d_in[mb + 7], t3);
        k_scan<<<256, 256, 0, stream>>>(t3, t1, t2, (const float*)d_in[mb + 8], t4);
        k_gate<<<(BL * 256 + 255) / 256, 256, 0, stream>>>(
            t4, t1, (const float*)d_in[mb + 9], t0, xbf);
        k_cvtw<<<(128 * 256 + 255) / 256, 256, 0, stream>>>(
            (const float*)d_in[mb + 10], wbuf, 128, 128, 256);
        k_gemm<<<dim3(125, 2), 256, 0, stream>>>(xbf, wbuf, h, nullptr,
                                                 BL, 128, 256, 128, 1);

        if ((i + 1) % 3 == 0) {
            const int ab = 71 + ai * 6; ai++;
            k_ln_bf16<<<2000, 256, 0, stream>>>(h, (const float*)d_in[ab + 0],
                                                (const float*)d_in[ab + 1], xbf, BL);
            k_cvtw<<<(384 * 128 + 255) / 256, 256, 0, stream>>>(
                (const float*)d_in[ab + 2], wbuf, 384, 384, 128);
            k_gemm<<<dim3(125, 6), 256, 0, stream>>>(xbf, wbuf, t0,
                (const float*)d_in[ab + 3], BL, 384, 128, 384, 0);
            k_attn<<<BSZ_ * NHEADS * (L_SEQ / 16), 32, 0, stream>>>(t0, t4);
            k_cvt<<<(BL * 128 + 255) / 256, 256, 0, stream>>>(t4, xbf, BL * 128);
            k_cvtw<<<(128 * 128 + 255) / 256, 256, 0, stream>>>(
                (const float*)d_in[ab + 4], wbuf, 128, 128, 128);
            k_gemm<<<dim3(125, 2), 256, 0, stream>>>(xbf, wbuf, h,
                (const float*)d_in[ab + 5], BL, 128, 128, 128, 1);
        }
    }

    k_final<<<2000, 256, 0, stream>>>(h, (const float*)d_in[83],
        (const float*)d_in[84], (const float*)d_in[85],
        (const float*)d_in[86], (float*)d_out, BL);
}
